// GIN_11751030522384
// MI455X (gfx1250) — matrix operand order repaired and checked by reading, not compiled, not run on hardware
//
#include <hip/hip_runtime.h>

typedef __attribute__((ext_vector_type(16))) _Float16 v16h;
typedef __attribute__((ext_vector_type(8)))  float    v8f;
typedef __attribute__((ext_vector_type(4)))  _Float16 half4;

#define N_NODES   50000
#define DIM       512
#define N_EDGES   400000
#define N_CLASSES 40

// ---------------------------------------------------------------- zero agg
__global__ void gin_zero_f4(float4* __restrict__ p, int n4) {
    int i = blockIdx.x * blockDim.x + threadIdx.x;
    if (i < n4) p[i] = make_float4(0.f, 0.f, 0.f, 0.f);
}

// ---------------------------------------------------------------- scatter-add
// One wave (32 lanes) per edge: agg[dst] += h[src] over 512 floats.
// h fits in L2 (102MB < 192MB) so gathers are L2 hits; non-returning
// relaxed agent-scope fadd -> global_atomic_add_f32 resolved in L2.
__global__ void gin_scatter_add(const float* __restrict__ h,
                                const int*   __restrict__ src,
                                const int*   __restrict__ dst,
                                float*       __restrict__ agg,
                                int n_edges) {
    int e = blockIdx.x * (blockDim.x >> 5) + (threadIdx.x >> 5);
    if (e >= n_edges) return;
    int lane = threadIdx.x & 31;
    size_t s = (size_t)src[e];
    size_t d = (size_t)dst[e];
    const float4* hp = (const float4*)(h + s * DIM);
    float* ap = agg + d * DIM;
#pragma unroll
    for (int j = 0; j < 4; ++j) {
        int col = j * 128 + lane * 4;      // coalesced 512B segments per pass
        float4 v = hp[col >> 2];
        __hip_atomic_fetch_add(ap + col + 0, v.x, __ATOMIC_RELAXED, __HIP_MEMORY_SCOPE_AGENT);
        __hip_atomic_fetch_add(ap + col + 1, v.y, __ATOMIC_RELAXED, __HIP_MEMORY_SCOPE_AGENT);
        __hip_atomic_fetch_add(ap + col + 2, v.z, __ATOMIC_RELAXED, __HIP_MEMORY_SCOPE_AGENT);
        __hip_atomic_fetch_add(ap + col + 3, v.w, __ATOMIC_RELAXED, __HIP_MEMORY_SCOPE_AGENT);
    }
}

// ---------------------------------------------------------------- GEMM tile staging
// A tile: 128 rows x 32 K of (h + agg), f32 -> f16, row-major, stride 36 halves.
// B tile: 32 K x 64 N of W, f32 -> f16, stored transposed sB[n][k] so that the
// (k, k+1) pair a B-fragment lane needs is one aligned 4-byte LDS load.
__device__ __forceinline__ void gin_stage_tile(
    _Float16 (* __restrict__ sA)[36], _Float16 (* __restrict__ sB)[36],
    const float* __restrict__ hin, const float* __restrict__ agg,
    const float* __restrict__ W,
    int tid, int m0, int n0, int kk, int nrows, int n_out) {
#pragma unroll
    for (int p = 0; p < 4; ++p) {               // 1024 float4-chunks of A
        int chunk = tid + (p << 8);
        int row   = chunk >> 3;
        int colc  = (chunk & 7) << 2;
        int rg    = m0 + row;
        float4 v  = make_float4(0.f, 0.f, 0.f, 0.f);
        if (rg < nrows) {
            v = *(const float4*)(hin + (size_t)rg * DIM + (kk + colc));
            if (agg) {
                float4 a = *(const float4*)(agg + (size_t)rg * DIM + (kk + colc));
                v.x += a.x; v.y += a.y; v.z += a.z; v.w += a.w;
            }
        }
        half4 hv = { (_Float16)v.x, (_Float16)v.y, (_Float16)v.z, (_Float16)v.w };
        *(half4*)&sA[row][colc] = hv;           // 8B aligned (stride 72B)
    }
#pragma unroll
    for (int p = 0; p < 8; ++p) {               // 2048 elements of B, coalesced in n
        int el = tid + (p << 8);
        int k  = el >> 6;
        int n  = el & 63;
        int ng = n0 + n;
        float w = (ng < n_out) ? W[(size_t)(kk + k) * n_out + ng] : 0.f;
        sB[n][k] = (_Float16)w;
    }
}

// ---------------------------------------------------------------- fused GEMM
// out[r][c] = act( (h+agg)[r][:] @ W[:][c] + bias[c] )
// Block: 256 thr = 8 waves. Block tile 128x64. Wave tile 32x32 = 2x2 WMMA frags.
__launch_bounds__(256)
__global__ void gin_gemm_bias_act(const float* __restrict__ hin,
                                  const float* __restrict__ agg,   // may be null
                                  const float* __restrict__ W,     // [DIM][n_out]
                                  const float* __restrict__ bias,  // [n_out]
                                  float*       __restrict__ out,   // [nrows][n_out]
                                  int nrows, int n_out, int do_relu) {
    __shared__ _Float16 sA[2][128][36];
    __shared__ _Float16 sB[2][64][36];

    const int tid   = threadIdx.x;
    const int lane  = tid & 31;
    const int wave  = tid >> 5;
    const int waveM = wave & 3;          // 4 strips of 32 rows
    const int waveN = wave >> 2;         // 2 strips of 32 cols
    const int m0    = blockIdx.x * 128;
    const int n0    = blockIdx.y * 64;
    const int m_loc = lane & 15;
    const int hsel  = lane >> 4;

    v8f acc[2][2] = {};

    const int NT = DIM / 32;             // 16 K-steps
    gin_stage_tile(sA[0], sB[0], hin, agg, W, tid, m0, n0, 0, nrows, n_out);

    // unroll 2: buffer index (t & 1) becomes a compile-time constant, so the
    // double-buffer base folds into ds_load immediate offsets instead of a
    // per-iteration v_mad, and the scheduler can pipeline two WMMA groups.
#pragma unroll 2
    for (int t = 0; t < NT; ++t) {
        __syncthreads();
        if (t + 1 < NT)
            gin_stage_tile(sA[(t + 1) & 1], sB[(t + 1) & 1], hin, agg, W,
                           tid, m0, n0, (t + 1) * 32, nrows, n_out);

        const int buf = t & 1;
        union Frag { unsigned u[8]; v16h h; };
        Frag aF[2], bF[2];
        // A fragment: 16x32 f16 per ISA layout. lanes 0-15: K {0..7,16..23},
        // lanes 16-31: K {8..15,24..31}; VGPR v holds pair (k,k+1).
        const int kbase = hsel << 3;
#pragma unroll
        for (int i = 0; i < 2; ++i) {
            const _Float16* rp = &sA[buf][waveM * 32 + i * 16 + m_loc][0];
#pragma unroll
            for (int v = 0; v < 8; ++v) {
                int k = (v < 4) ? (2 * v + kbase) : (16 + 2 * (v - 4) + kbase);
                aF[i].u[v] = *(const unsigned*)(rp + k);
            }
        }
        // B fragment: 32x16 f16. lane&15 = column n; lanes 0-15: K {0..7,16..23},
        // lanes 16-31: K {8..15,24..31} (same order as A).
        const int kb = hsel << 3;
#pragma unroll
        for (int j = 0; j < 2; ++j) {
            const _Float16* rp = &sB[buf][waveN * 32 + j * 16 + m_loc][0];
#pragma unroll
            for (int v = 0; v < 8; ++v)
                bF[j].u[v] = *(const unsigned*)(rp + kb + ((v < 4) ? (2 * v) : (16 + 2 * (v - 4))));
        }
#pragma unroll
        for (int i = 0; i < 2; ++i)
#pragma unroll
            for (int j = 0; j < 2; ++j)
                acc[i][j] = __builtin_amdgcn_wmma_f32_16x16x32_f16(
                    false, aF[i].h, false, bF[j].h,
                    (short)0, acc[i][j], false, false);
    }

    // epilogue: C layout -> VGPR r is row (r + 8*hsel), col = lane&15
    float bv[2];
#pragma unroll
    for (int j = 0; j < 2; ++j) {
        int cg = n0 + waveN * 32 + j * 16 + m_loc;
        bv[j] = (cg < n_out) ? bias[cg] : 0.f;
    }
#pragma unroll
    for (int i = 0; i < 2; ++i) {
        int rbase = m0 + waveM * 32 + i * 16 + hsel * 8;
#pragma unroll
        for (int j = 0; j < 2; ++j) {
            int cg = n0 + waveN * 32 + j * 16 + m_loc;
#pragma unroll
            for (int r = 0; r < 8; ++r) {
                int rg = rbase + r;
                if (rg < nrows && cg < n_out) {
                    float o = acc[i][j][r] + bv[j];
                    if (do_relu) o = fmaxf(o, 0.f);
                    out[(size_t)rg * n_out + cg] = o;
                }
            }
        }
    }
}

// ---------------------------------------------------------------- launch
extern "C" void kernel_launch(void* const* d_in, const int* in_sizes, int n_in,
                              void* d_out, int out_size, void* d_ws, size_t ws_size,
                              hipStream_t stream) {
    const float* feat = (const float*)d_in[0];
    const int*   src  = (const int*)  d_in[1];
    const int*   dst  = (const int*)  d_in[2];
    const float* W0   = (const float*)d_in[3];
    const float* b0   = (const float*)d_in[4];
    const float* W1   = (const float*)d_in[5];
    const float* b1   = (const float*)d_in[6];
    const float* W2   = (const float*)d_in[7];
    const float* b2   = (const float*)d_in[8];
    const float* Wc   = (const float*)d_in[9];
    const float* bc   = (const float*)d_in[10];
    float* out = (float*)d_out;

    const size_t nd = (size_t)N_NODES * DIM;
    float* hA  = (float*)d_ws;       // 102.4 MB
    float* hB  = hA + nd;            // 102.4 MB
    float* agg = hB + nd;            // 102.4 MB

    const int n4 = (int)(nd / 4);
    dim3 blk(256);
    dim3 zgrid((n4 + 255) / 256);
    dim3 sgrid((N_EDGES + 7) / 8);
    dim3 ggrid((N_NODES + 127) / 128, DIM / 64);
    dim3 cgrid((N_NODES + 127) / 128, 1);

    // layer 0: feat -> hA
    gin_zero_f4<<<zgrid, blk, 0, stream>>>((float4*)agg, n4);
    gin_scatter_add<<<sgrid, blk, 0, stream>>>(feat, src, dst, agg, N_EDGES);
    gin_gemm_bias_act<<<ggrid, blk, 0, stream>>>(feat, agg, W0, b0, hA,
                                                 N_NODES, DIM, 1);
    // layer 1: hA -> hB
    gin_zero_f4<<<zgrid, blk, 0, stream>>>((float4*)agg, n4);
    gin_scatter_add<<<sgrid, blk, 0, stream>>>(hA, src, dst, agg, N_EDGES);
    gin_gemm_bias_act<<<ggrid, blk, 0, stream>>>(hA, agg, W1, b1, hB,
                                                 N_NODES, DIM, 1);
    // layer 2: hB -> hA
    gin_zero_f4<<<zgrid, blk, 0, stream>>>((float4*)agg, n4);
    gin_scatter_add<<<sgrid, blk, 0, stream>>>(hB, src, dst, agg, N_EDGES);
    gin_gemm_bias_act<<<ggrid, blk, 0, stream>>>(hB, agg, W2, b2, hA,
                                                 N_NODES, DIM, 1);
    // classifier: hA -> out (no agg, no relu)
    gin_gemm_bias_act<<<cgrid, blk, 0, stream>>>(hA, nullptr, Wc, bc, out,
                                                 N_NODES, N_CLASSES, 0);
}
